// Fp8Linear_4844723110281
// MI455X (gfx1250) — 8-bit matrix operand order repaired and checked by reading, compiled, not run on hardware
//
#include <hip/hip_runtime.h>
#include <stdint.h>

typedef __attribute__((ext_vector_type(16))) int          v16i;
typedef __attribute__((ext_vector_type(8)))  float        v8f;
typedef __attribute__((ext_vector_type(4)))  unsigned int u32x4;
typedef __attribute__((ext_vector_type(4)))  float        f32x4;
typedef int vs4i __attribute__((vector_size(16)));
typedef unsigned long long u64;

#define AS1 __attribute__((address_space(1)))
#define AS3 __attribute__((address_space(3)))

#define TILE 128

#if __has_builtin(__builtin_amdgcn_global_load_async_to_lds_b128)
#define HAVE_ASYNC_LDS 1
#else
#define HAVE_ASYNC_LDS 0
#endif

__device__ __forceinline__ void wait_async_zero() {
#if __has_builtin(__builtin_amdgcn_s_wait_asynccnt)
    __builtin_amdgcn_s_wait_asynccnt(0);
#else
    asm volatile("s_wait_asynccnt 0x0" ::: "memory");
#endif
}

#if !__has_builtin(__builtin_amdgcn_cvt_pk_fp8_f32)
__device__ __forceinline__ uint32_t f32_to_e4m3(float f) {
    uint32_t sign = (__float_as_uint(f) >> 24) & 0x80u;
    float a = fabsf(f);
    if (!(a > 2.9802322e-10f)) return sign;
    if (a > 448.f) a = 448.f;
    int e;
    float m = frexpf(a, &e);
    if (e <= -6) {
        int mant = (int)rintf(a * 512.f);
        return sign | (uint32_t)mant;
    }
    int q = (int)rintf(m * 16.f);
    if (q == 16) { q = 8; ++e; }
    return sign | ((uint32_t)(e + 6) << 3) | (uint32_t)(q - 8);
}
#endif

__device__ __forceinline__ uint32_t pack4_e4m3(f32x4 v) {
#if __has_builtin(__builtin_amdgcn_cvt_pk_fp8_f32)
    int p = __builtin_amdgcn_cvt_pk_fp8_f32(v.x, v.y, 0, false);
    p = __builtin_amdgcn_cvt_pk_fp8_f32(v.z, v.w, p, true);
    return (uint32_t)p;
#else
    return f32_to_e4m3(v.x) | (f32_to_e4m3(v.y) << 8) |
           (f32_to_e4m3(v.z) << 16) | (f32_to_e4m3(v.w) << 24);
#endif
}

__global__ __launch_bounds__(256) void quant_rows_kernel(
    const float* __restrict__ src, uint8_t* __restrict__ q,
    float* __restrict__ scales, int K) {
    const int row = blockIdx.x;
    const int tid = threadIdx.x;
    const f32x4* s4 = (const f32x4*)(src + (size_t)row * K);
    uint32_t* q4 = (uint32_t*)(q + (size_t)row * K);
    const int n4 = K >> 2;

    __shared__ f32x4 rowc[1024];
    __shared__ float red[256];

    float amax = 0.f;
    const bool cached = (n4 <= 1024);
    for (int i = tid; i < n4; i += 256) {
        f32x4 v = __builtin_nontemporal_load(&s4[i]);
        if (cached) rowc[i] = v;
        amax = fmaxf(amax, fmaxf(fmaxf(fabsf(v.x), fabsf(v.y)),
                                 fmaxf(fabsf(v.z), fabsf(v.w))));
    }
    red[tid] = amax;
    __syncthreads();
    for (int s = 128; s > 0; s >>= 1) {
        if (tid < s) red[tid] = fmaxf(red[tid], red[tid + s]);
        __syncthreads();
    }
    float scale = fmaxf(red[0] * (1.f / 448.f), 1e-12f);
    if (tid == 0) scales[row] = scale;
    const float inv = 1.f / scale;
    for (int i = tid; i < n4; i += 256) {
        f32x4 v = cached ? rowc[i] : __builtin_nontemporal_load(&s4[i]);
        f32x4 s = {v.x * inv, v.y * inv, v.z * inv, v.w * inv};
        q4[i] = pack4_e4m3(s);
    }
}

__global__ __launch_bounds__(256, 1) void fp8_gemm_kernel(
    const uint8_t* __restrict__ Xq, const uint8_t* __restrict__ Wq,
    const float* __restrict__ xscale, const float* __restrict__ wscale,
    const float* __restrict__ bias, float* __restrict__ out,
    int M, int N, int K) {
    __shared__ uint8_t lA[2][TILE * TILE];
    __shared__ uint8_t lB[2][TILE * TILE];

    const int tid  = threadIdx.x;
    const int lane = tid & 31;
    const int wave = tid >> 5;
    const int wm   = wave & 3;
    const int wn   = wave >> 2;
    const int hi   = lane >> 4;
    const int l16  = lane & 15;

    const int bm = blockIdx.y * TILE;
    const int bn = blockIdx.x * TILE;

    v8f acc[2][4] = {};

    auto load_tile = [&](int kt, int buf) {
        const size_t k0 = (size_t)kt * TILE;
#pragma unroll
        for (int i = 0; i < 4; ++i) {
            int c   = tid + i * 256;
            int r   = c >> 3;
            int col = (c & 7) << 4;
#if HAVE_ASYNC_LDS
            __builtin_amdgcn_global_load_async_to_lds_b128(
                (AS1 vs4i*)(Xq + (size_t)(bm + r) * K + k0 + col),
                (AS3 vs4i*)&lA[buf][r * TILE + col], 0, 0);
            __builtin_amdgcn_global_load_async_to_lds_b128(
                (AS1 vs4i*)(Wq + (size_t)(bn + r) * K + k0 + col),
                (AS3 vs4i*)&lB[buf][r * TILE + col], 0, 0);
#else
            *(u32x4*)&lA[buf][r * TILE + col] =
                *(const u32x4*)&Xq[(size_t)(bm + r) * K + k0 + col];
            *(u32x4*)&lB[buf][r * TILE + col] =
                *(const u32x4*)&Wq[(size_t)(bn + r) * K + k0 + col];
#endif
        }
    };

    load_tile(0, 0);
#if HAVE_ASYNC_LDS
    wait_async_zero();
#endif
    __syncthreads();

    const int KT = K / TILE;
    for (int kt = 0; kt < KT; ++kt) {
        const int cur = kt & 1;
        if (kt + 1 < KT) load_tile(kt + 1, cur ^ 1);

        union { v16i v; u64 q[8]; } af[2];
#pragma unroll
        for (int tm = 0; tm < 2; ++tm) {
            const uint8_t* base =
                &lA[cur][(wm * 32 + tm * 16 + l16) * TILE + hi * 16];
#pragma unroll
            for (int j = 0; j < 8; ++j)
                af[tm].q[j] = *(const u64*)(base + (j >> 1) * 32 + (j & 1) * 8);
        }

#pragma unroll
        for (int tn = 0; tn < 4; ++tn) {
            union { v16i v; u32x4 q[4]; } bf;
            const uint8_t* baseB =
                &lB[cur][(wn * 64 + tn * 16 + l16) * TILE + hi * 16];
#pragma unroll
            for (int g = 0; g < 4; ++g)
                bf.q[g] = *(const u32x4*)(baseB + g * 32);
#pragma unroll
            for (int tm = 0; tm < 2; ++tm)
                acc[tm][tn] = __builtin_amdgcn_wmma_f32_16x16x128_fp8_fp8(
                    af[tm].v, bf.v, (short)0, acc[tm][tn],
                     false,  false);
        }

#if HAVE_ASYNC_LDS
        wait_async_zero();
#endif
        __syncthreads();
    }

#pragma unroll
    for (int tn = 0; tn < 4; ++tn) {
        const int n   = bn + wn * 64 + tn * 16 + l16;
        const float w = wscale[n];
        const float b = bias[n];
#pragma unroll
        for (int tm = 0; tm < 2; ++tm) {
            const int mbase = bm + wm * 32 + tm * 16 + hi * 8;
#pragma unroll
            for (int r = 0; r < 8; ++r) {
                const int m = mbase + r;
                __builtin_nontemporal_store(acc[tm][tn][r] * xscale[m] * w + b,
                                            &out[(size_t)m * N + n]);
            }
        }
    }
}

extern "C" void kernel_launch(void* const* d_in, const int* in_sizes, int n_in,
                              void* d_out, int out_size, void* d_ws, size_t ws_size,
                              hipStream_t stream) {
    const float* x    = (const float*)d_in[0];
    const float* w    = (const float*)d_in[1];
    const float* bias = (const float*)d_in[2];
    float* out        = (float*)d_out;

    const int N = in_sizes[2];
    const int K = in_sizes[1] / N;
    const int M = in_sizes[0] / K;

    uint8_t* ws8 = (uint8_t*)d_ws;
    uint8_t* xq  = ws8;
    uint8_t* wq  = ws8 + (size_t)M * K;
    float* xs    = (float*)(ws8 + (size_t)M * K + (size_t)N * K);
    float* wsc   = xs + M;

    quant_rows_kernel<<<M, 256, 0, stream>>>(x, xq, xs, K);
    quant_rows_kernel<<<N, 256, 0, stream>>>(w, wq, wsc, K);

    dim3 grid(N / TILE, M / TILE);
    fp8_gemm_kernel<<<grid, 256, 0, stream>>>(xq, wq, xs, wsc, bias, out,
                                              M, N, K);
}
